// TransformerDecoderBlock_72318659330259
// MI455X (gfx1250) — compile-verified, matrix operand order checked by reading, not run on hardware
//
#include <hip/hip_runtime.h>
#include <hip/hip_bf16.h>
#include <stdint.h>

// ---------------------------------------------------------------------------
// Problem constants: B=4 S=1024 M=1024 D=1024 H=16 DH=64 F=4096 E=4 K=2
// ---------------------------------------------------------------------------
#define DIM_B 4
#define DIM_S 1024
#define DIM_D 1024
#define DIM_H 16
#define DIM_DH 64
#define DIM_F 4096
#define DIM_E 4
#define NROWS (DIM_B * DIM_S)        // 4096 token rows
#define LN_EPS 1e-5f

typedef __attribute__((ext_vector_type(16))) __bf16 v16bf;
typedef __attribute__((ext_vector_type(8)))  float  v8f;
typedef __attribute__((ext_vector_type(4)))  int    v4i;

struct __align__(16) U4 { unsigned int x, y, z, w; };
union Frag16 { v16bf v; U4 q[2]; };

// Async global->LDS path (gfx1250). Guarded so compilation never regresses.
#if defined(__has_builtin)
#  if __has_builtin(__builtin_amdgcn_global_load_async_to_lds_b128)
#    define ASYNC_EN 1
#  endif
#endif
#ifndef ASYNC_EN
#  define ASYNC_EN 0
#endif

__device__ __forceinline__ unsigned short f2bf(float f) {
    union { float f; uint32_t u; } c; c.f = f;
    uint32_t u = c.u;
    return (unsigned short)((u + 0x7FFFu + ((u >> 16) & 1u)) >> 16);  // RNE
}
__device__ __forceinline__ float bf2f(unsigned short s) {
    union { uint32_t u; float f; } c; c.u = ((uint32_t)s) << 16; return c.f;
}
__device__ __forceinline__ float gelu_exact(float x) {
    return 0.5f * x * (1.0f + erff(x * 0.70710678118654752f));
}

// 16-byte tile chunk: global (bf16) -> LDS, asynchronously when available.
// Builtin signature (from hipcc diagnostics): (v4i AS1*, v4i AS3*, imm, imm)
__device__ __forceinline__ void copy16(unsigned short* ldsDst, const unsigned short* gSrc) {
#if ASYNC_EN
    __builtin_amdgcn_global_load_async_to_lds_b128(
        (__attribute__((address_space(1))) v4i*)gSrc,
        (__attribute__((address_space(3))) v4i*)ldsDst,
        0, 0);
#else
    *(U4*)ldsDst = *(const U4*)gSrc;
#endif
}

// ---------------------------------------------------------------------------
// bf16 WMMA GEMM:  C[r][c] = sum_k A[r][k] * Bm[c][k]   (all our GEMMs are
// x @ W^T / Q @ K^T / P @ Vt^T form; both operands K-contiguous in memory).
// Template: BN = 128 or 64 (N tile), EPI: 0 none | 1 bias+gelu | 2 bias |
// 3 residual-add; OUTBF: 0 fp32 out | 1 bf16 out.
// 256 threads = 8 waves; M tile 128; per-wave 32 x (BN/2); K-step 32.
// Double-buffered LDS filled by GLOBAL_LOAD_ASYNC_TO_LDS_B128, consumed via
// ds_load_b128 fragments -> v_wmma_f32_16x16x32_bf16.
// No bounds guards: every launch is an exact multiple of the tile.
// ---------------------------------------------------------------------------
template <int BN, int EPI, int OUTBF>
__global__ __launch_bounds__(256)
void gemm_kernel(const unsigned short* __restrict__ A,  int lda, long aOffB, long aOffH, long aOffZ,
                 const unsigned short* __restrict__ Bm, int ldb, long bOffB, long bOffH, long bOffZ,
                 void* __restrict__ Cv,                 int ldc, long cOffB, long cOffH, long cOffZ,
                 const float* __restrict__ bias,
                 const float* __restrict__ res, int ldres,
                 int n_k, int hDiv, int zBase)
{
    constexpr int NT  = BN / 32;             // N sub-tiles per wave (4 or 2)
    constexpr int ACH = 2;                   // A 16B-chunks per thread
    constexpr int BCH = BN * 4 / 256;        // B 16B-chunks per thread (2 or 1)
    constexpr int PER_TILE = ACH + BCH;      // async ops in flight per tile

    __shared__ __align__(16) unsigned short shA[2][128 * 40];
    __shared__ __align__(16) unsigned short shB[2][BN * 40];

    const int zl = blockIdx.z;
    const int zg = zBase + zl;
    const int bb = zg / hDiv;
    const int hh = zg % hDiv;
    const unsigned short* Ap = A  + (long)bb * aOffB + (long)hh * aOffH + (long)zl * aOffZ;
    const unsigned short* Bp = Bm + (long)bb * bOffB + (long)hh * bOffH + (long)zl * bOffZ;

    const int rowBase = blockIdx.y * 128;
    const int colBase = blockIdx.x * BN;
    const int t    = threadIdx.x;
    const int lane = t & 31;
    const int wid  = t >> 5;
    const int wm   = wid & 3;                // 4 wave-rows * 32 = 128
    const int wn   = wid >> 2;               // 2 wave-cols * (BN/2)
    const int lo   = lane & 15;
    const int hi   = lane >> 4;

    v8f acc[2][NT];
#pragma unroll
    for (int mt = 0; mt < 2; ++mt)
#pragma unroll
        for (int nt = 0; nt < NT; ++nt) { v8f z = {}; acc[mt][nt] = z; }

    auto issueTile = [&](int buf, int k0) {
#pragma unroll
        for (int i = 0; i < ACH; ++i) {
            int flat = t + i * 256;          // 512 chunks: 128 rows x 4
            int r = flat >> 2, ch = flat & 3;
            copy16(&shA[buf][r * 40 + ch * 8],
                   Ap + (long)(rowBase + r) * lda + k0 + ch * 8);
        }
#pragma unroll
        for (int i = 0; i < BCH; ++i) {
            int flat = t + i * 256;          // BN*4 chunks
            int c = flat >> 2, ch = flat & 3;
            copy16(&shB[buf][c * 40 + ch * 8],
                   Bp + (long)(colBase + c) * ldb + k0 + ch * 8);
        }
    };

    int cur = 0;
    issueTile(0, 0);
    for (int k0 = 0; k0 < n_k; k0 += 32) {
        const bool hasNext = (k0 + 32) < n_k;
        if (hasNext) issueTile(cur ^ 1, k0 + 32);
#if ASYNC_EN
        if (hasNext) asm volatile("s_wait_asynccnt %0" :: "i"(PER_TILE) : "memory");
        else         asm volatile("s_wait_asynccnt 0"  ::: "memory");
#endif
        __syncthreads();

        const unsigned short* sA = shA[cur];
        const unsigned short* sB = shB[cur];
        Frag16 fa[2], fb[NT];
#pragma unroll
        for (int mt = 0; mt < 2; ++mt) {
            const unsigned short* p = &sA[(wm * 32 + mt * 16 + lo) * 40 + hi * 8];
            fa[mt].q[0] = *(const U4*)p;
            fa[mt].q[1] = *(const U4*)(p + 16);
        }
#pragma unroll
        for (int nt = 0; nt < NT; ++nt) {
            const unsigned short* p = &sB[(wn * (NT * 16) + nt * 16 + lo) * 40 + hi * 8];
            fb[nt].q[0] = *(const U4*)p;
            fb[nt].q[1] = *(const U4*)(p + 16);
        }
#pragma unroll
        for (int mt = 0; mt < 2; ++mt)
#pragma unroll
            for (int nt = 0; nt < NT; ++nt)
                acc[mt][nt] = __builtin_amdgcn_wmma_f32_16x16x32_bf16(
                    false, fa[mt].v, false, fb[nt].v,
                    (short)0, acc[mt][nt], false, false);
        __syncthreads();                     // reads done before buffer reuse
        cur ^= 1;
    }

    // epilogue + store (C/D layout: vgpr j -> M = j + 8*hi, N = lo)
    float*          Cf = (float*)Cv          + (long)bb * cOffB + (long)hh * cOffH + (long)zl * cOffZ;
    unsigned short* Cb = (unsigned short*)Cv + (long)bb * cOffB + (long)hh * cOffH + (long)zl * cOffZ;
#pragma unroll
    for (int mt = 0; mt < 2; ++mt) {
#pragma unroll
        for (int nt = 0; nt < NT; ++nt) {
            const int gc = colBase + wn * (NT * 16) + nt * 16 + lo;
#pragma unroll
            for (int j = 0; j < 8; ++j) {
                const int gr = rowBase + wm * 32 + mt * 16 + j + hi * 8;
                float v = acc[mt][nt][j];
                if (EPI == 1) { v += bias[gc]; v = gelu_exact(v); }
                else if (EPI == 2) { v += bias[gc]; }
                else if (EPI == 3) { v += res[(long)gr * ldres + gc]; }
                if (OUTBF) Cb[(long)gr * ldc + gc] = f2bf(v);
                else       Cf[(long)gr * ldc + gc] = v;
            }
        }
    }
}

// ---------------------------------------------------------------------------
// fp32 -> bf16 elementwise conversion (weights / memory, once per call)
// ---------------------------------------------------------------------------
__global__ __launch_bounds__(256)
void cvt_kernel(const float* __restrict__ src, unsigned short* __restrict__ dst)
{
    long i = (long)blockIdx.x * 256 + threadIdx.x;
    dst[i] = f2bf(src[i]);
}

// ---------------------------------------------------------------------------
// LayerNorm over D=1024 -> bf16 output (GEMM operand). One block per row.
// ---------------------------------------------------------------------------
__global__ __launch_bounds__(256)
void ln_kernel(const float* __restrict__ x, const float* __restrict__ g,
               const float* __restrict__ b, unsigned short* __restrict__ out)
{
    const long r = blockIdx.x;
    const int  t = threadIdx.x;
    const float* xr = x + r * DIM_D;
    float v[4], ls = 0.f, lq = 0.f;
#pragma unroll
    for (int i = 0; i < 4; ++i) { v[i] = xr[t + i * 256]; ls += v[i]; lq += v[i] * v[i]; }
    __shared__ float s1[256], s2[256];
    s1[t] = ls; s2[t] = lq; __syncthreads();
    for (int o = 128; o > 0; o >>= 1) {
        if (t < o) { s1[t] += s1[t + o]; s2[t] += s2[t + o]; }
        __syncthreads();
    }
    const float mean = s1[0] * (1.f / DIM_D);
    const float var  = s2[0] * (1.f / DIM_D) - mean * mean;
    const float rstd = rsqrtf(var + LN_EPS);
#pragma unroll
    for (int i = 0; i < 4; ++i) {
        int d = t + i * 256;
        out[r * DIM_D + d] = f2bf((v[i] - mean) * rstd * g[d] + b[d]);
    }
}

// ---------------------------------------------------------------------------
// Row softmax (len 1024) with pre-scale: fp32 scores in, bf16 probs out.
// ---------------------------------------------------------------------------
__global__ __launch_bounds__(256)
void softmax_kernel(const float* __restrict__ Pin, unsigned short* __restrict__ Pout, float scale)
{
    const long r = blockIdx.x;
    const float* row = Pin + r * 1024;
    unsigned short* orow = Pout + r * 1024;
    const int t = threadIdx.x;
    __shared__ float s[256];
    float v[4], m = -3.4e38f;
#pragma unroll
    for (int i = 0; i < 4; ++i) { v[i] = row[t + i * 256] * scale; m = fmaxf(m, v[i]); }
    s[t] = m; __syncthreads();
    for (int o = 128; o > 0; o >>= 1) { if (t < o) s[t] = fmaxf(s[t], s[t + o]); __syncthreads(); }
    const float mx = s[0]; __syncthreads();
    float sum = 0.f;
#pragma unroll
    for (int i = 0; i < 4; ++i) { v[i] = expf(v[i] - mx); sum += v[i]; }
    s[t] = sum; __syncthreads();
    for (int o = 128; o > 0; o >>= 1) { if (t < o) s[t] += s[t + o]; __syncthreads(); }
    const float inv = 1.f / s[0];
#pragma unroll
    for (int i = 0; i < 4; ++i) orow[t + i * 256] = f2bf(v[i] * inv);
}

// ---------------------------------------------------------------------------
// V [B*S, D] bf16  ->  Vt [B*H, DH, S] bf16 so attn@V is A@B^T like the rest.
// ---------------------------------------------------------------------------
__global__ __launch_bounds__(256)
void transpose_v_kernel(const unsigned short* __restrict__ V, unsigned short* __restrict__ Vt)
{
    long i = (long)blockIdx.x * 256 + threadIdx.x;     // 4M elements
    int k  = (int)(i & 1023);
    int d  = (int)((i >> 10) & 63);
    int bh = (int)(i >> 16);
    int b = bh >> 4, h = bh & 15;
    Vt[i] = V[((long)(b * 1024 + k)) * 1024 + h * 64 + d];
}

// ---------------------------------------------------------------------------
// MoE gate: logits = h @ Wg^T (E=4); softmax probs; top-2 + renorm weights.
// ---------------------------------------------------------------------------
__global__ __launch_bounds__(256)
void gate_kernel(const unsigned short* __restrict__ h, const float* __restrict__ Wg,
                 float* __restrict__ probs, float* __restrict__ tw, int* __restrict__ tidx)
{
    const long r = blockIdx.x;
    const int  t = threadIdx.x;
    float a[4] = {0.f, 0.f, 0.f, 0.f};
#pragma unroll
    for (int i = 0; i < 4; ++i) {
        int d = t + i * 256;
        float hv = bf2f(h[r * DIM_D + d]);
#pragma unroll
        for (int e = 0; e < 4; ++e) a[e] += hv * Wg[e * DIM_D + d];
    }
    __shared__ float s[4][256];
#pragma unroll
    for (int e = 0; e < 4; ++e) s[e][t] = a[e];
    __syncthreads();
    for (int o = 128; o > 0; o >>= 1) {
        if (t < o) {
#pragma unroll
            for (int e = 0; e < 4; ++e) s[e][t] += s[e][t + o];
        }
        __syncthreads();
    }
    if (t == 0) {
        float lg[4] = { s[0][0], s[1][0], s[2][0], s[3][0] };
        float mx = fmaxf(fmaxf(lg[0], lg[1]), fmaxf(lg[2], lg[3]));
        float ex[4], sum = 0.f;
#pragma unroll
        for (int e = 0; e < 4; ++e) { ex[e] = expf(lg[e] - mx); sum += ex[e]; }
#pragma unroll
        for (int e = 0; e < 4; ++e) probs[r * 4 + e] = ex[e] / sum;
        int i0 = 0;
#pragma unroll
        for (int e = 1; e < 4; ++e) if (lg[e] > lg[i0]) i0 = e;
        int i1 = -1;
#pragma unroll
        for (int e = 0; e < 4; ++e)
            if (e != i0 && (i1 < 0 || lg[e] > lg[i1])) i1 = e;
        float m2 = fmaxf(lg[i0], lg[i1]);
        float e0 = expf(lg[i0] - m2), e1 = expf(lg[i1] - m2);
        float ss = e0 + e1;
        tw[r * 2 + 0] = e0 / ss; tw[r * 2 + 1] = e1 / ss;
        tidx[r * 2 + 0] = i0;    tidx[r * 2 + 1] = i1;
    }
}

__global__ __launch_bounds__(256)
void aux_kernel(const float* __restrict__ probs, const int* __restrict__ tidx,
                float* __restrict__ auxOut)
{
    const int t = threadIdx.x;
    float imp[4] = {0.f, 0.f, 0.f, 0.f}, cnt[4] = {0.f, 0.f, 0.f, 0.f};
    for (int r = t; r < NROWS; r += 256) {
#pragma unroll
        for (int e = 0; e < 4; ++e) imp[e] += probs[(long)r * 4 + e];
        cnt[tidx[r * 2 + 0]] += 1.f;
        cnt[tidx[r * 2 + 1]] += 1.f;
    }
    __shared__ float si[4][256], sc[4][256];
#pragma unroll
    for (int e = 0; e < 4; ++e) { si[e][t] = imp[e]; sc[e][t] = cnt[e]; }
    __syncthreads();
    for (int o = 128; o > 0; o >>= 1) {
        if (t < o) {
#pragma unroll
            for (int e = 0; e < 4; ++e) { si[e][t] += si[e][t + o]; sc[e][t] += sc[e][t + o]; }
        }
        __syncthreads();
    }
    if (t == 0) {
        float aux = 0.f;
#pragma unroll
        for (int e = 0; e < 4; ++e)
            aux += (sc[e][0] / (float)NROWS) * (si[e][0] / (float)NROWS);
        auxOut[0] = 0.01f * (float)DIM_E * aux;
    }
}

__global__ __launch_bounds__(256)
void copy_kernel(const float* __restrict__ src, float* __restrict__ dst)
{
    long i = (long)blockIdx.x * 256 + threadIdx.x;
    dst[i] = src[i];
}

__global__ __launch_bounds__(256)
void moe_combine_kernel(const float* __restrict__ eo, const float* __restrict__ tw,
                        const int* __restrict__ tidx, float* __restrict__ out, int expert)
{
    long i = (long)blockIdx.x * 256 + threadIdx.x;
    long r = i >> 10;
    float w = 0.f;
    if (tidx[r * 2 + 0] == expert)      w = tw[r * 2 + 0];
    else if (tidx[r * 2 + 1] == expert) w = tw[r * 2 + 1];
    out[i] += w * eo[i];
}

// ---------------------------------------------------------------------------
// Host side
// ---------------------------------------------------------------------------
extern "C" void kernel_launch(void* const* d_in, const int* in_sizes, int n_in,
                              void* d_out, int out_size, void* d_ws, size_t ws_size,
                              hipStream_t stream)
{
    (void)in_sizes; (void)n_in; (void)out_size; (void)ws_size;
    const float* x    = (const float*)d_in[0];
    const float* memo = (const float*)d_in[1];
    const float* attW[8] = { (const float*)d_in[2], (const float*)d_in[3],
                             (const float*)d_in[4], (const float*)d_in[5],
                             (const float*)d_in[6], (const float*)d_in[7],
                             (const float*)d_in[8], (const float*)d_in[9] };
    const float* Wg   = (const float*)d_in[10];
    const float* W1   = (const float*)d_in[11];
    const float* b1   = (const float*)d_in[12];
    const float* W2   = (const float*)d_in[13];
    const float* b2   = (const float*)d_in[14];
    const float* ln1g = (const float*)d_in[15];
    const float* ln1b = (const float*)d_in[16];
    const float* ln2g = (const float*)d_in[17];
    const float* ln2b = (const float*)d_in[18];
    const float* ln3g = (const float*)d_in[19];
    const float* ln3b = (const float*)d_in[20];
    float* out = (float*)d_out;

    const size_t TOK = (size_t)NROWS * DIM_D;          // 4194304
    const size_t SS  = (size_t)DIM_S * DIM_S;          // 1048576

    // bump allocator over the workspace (256B aligned slices)
    char* wp = (char*)d_ws;
    auto alloc = [&](size_t bytes) { char* p = wp; wp += (bytes + 255) & ~(size_t)255; return p; };
    float* x1   = (float*)alloc(TOK * 4);
    float* x2   = (float*)alloc(TOK * 4);
    float* Pb   = (float*)alloc(16 * SS * 4);          // fp32 scores, 16-z chunk
    float* eo   = (float*)alloc(TOK * 4);
    unsigned short* hBf  = (unsigned short*)alloc(TOK * 2);
    unsigned short* Qbf  = (unsigned short*)alloc(TOK * 2);
    unsigned short* Kbf  = (unsigned short*)alloc(TOK * 2);
    unsigned short* Vbf  = (unsigned short*)alloc(TOK * 2);
    unsigned short* Vtbf = (unsigned short*)alloc(TOK * 2);
    unsigned short* AObf = (unsigned short*)alloc(TOK * 2);
    unsigned short* Pbf  = (unsigned short*)alloc(16 * SS * 2);
    unsigned short* hidBf = (unsigned short*)alloc((size_t)NROWS * DIM_F * 2);
    unsigned short* memBf = (unsigned short*)alloc(TOK * 2);
    unsigned short* wBf[8];
    for (int i = 0; i < 8; ++i) wBf[i] = (unsigned short*)alloc((size_t)DIM_D * DIM_D * 2);
    unsigned short* w1Bf = (unsigned short*)alloc((size_t)DIM_E * DIM_F * DIM_D * 2);
    unsigned short* w2Bf = (unsigned short*)alloc((size_t)DIM_E * DIM_D * DIM_F * 2);
    float* prb = (float*)alloc((size_t)NROWS * DIM_E * 4);
    float* tw  = (float*)alloc((size_t)NROWS * 2 * 4);
    int*   tidx = (int*)alloc((size_t)NROWS * 2 * 4);

    const dim3 blk(256);
    const dim3 gProj(DIM_D / 128, NROWS / 128, 1);     // (8, 32)
    const long SD = (long)DIM_S * DIM_D;

    auto cvt = [&](const float* s, unsigned short* d, size_t n) {
        cvt_kernel<<<(unsigned)(n / 256), blk, 0, stream>>>(s, d);
    };

    // ---- one-time (per call) operand conversions to bf16 ----
    for (int i = 0; i < 8; ++i) cvt(attW[i], wBf[i], (size_t)DIM_D * DIM_D);
    cvt(W1, w1Bf, (size_t)DIM_E * DIM_F * DIM_D);
    cvt(W2, w2Bf, (size_t)DIM_E * DIM_D * DIM_F);
    cvt(memo, memBf, TOK);

    // attention: Q/K/V already projected into Qbf/Kbf/Vbf (bf16, [B*S, D])
    auto run_attention = [&](const unsigned short* woBf, const float* resid, float* xout) {
        transpose_v_kernel<<<(unsigned)(TOK / 256), blk, 0, stream>>>(Vbf, Vtbf);
        for (int zb = 0; zb < DIM_B * DIM_H; zb += 16) {
            // scores = Q @ K^T over 16 (b,h) slices -> fp32
            gemm_kernel<128, 0, 0><<<dim3(8, 8, 16), blk, 0, stream>>>(
                Qbf, DIM_D, SD, DIM_DH, 0,
                Kbf, DIM_D, SD, DIM_DH, 0,
                Pb,  DIM_S, 0, 0, (long)SS,
                nullptr, nullptr, 0,
                DIM_DH, DIM_H, zb);
            // softmax (scale 1/sqrt(64)) -> bf16 probs
            softmax_kernel<<<16 * DIM_S, blk, 0, stream>>>(Pb, Pbf, 0.125f);
            // attn @ V == P @ Vt^T, BN=64, bf16 out into AO[b, q, h*64 + d]
            gemm_kernel<64, 0, 1><<<dim3(1, 8, 16), blk, 0, stream>>>(
                Pbf,  DIM_S, 0, 0, (long)SS,
                Vtbf, DIM_S, (long)DIM_H * DIM_DH * DIM_S, (long)DIM_DH * DIM_S, 0,
                AObf, DIM_D, SD, DIM_DH, 0,
                nullptr, nullptr, 0,
                DIM_S, DIM_H, zb);
        }
        // output projection + residual -> fp32
        gemm_kernel<128, 3, 0><<<gProj, blk, 0, stream>>>(
            AObf, DIM_D, 0, 0, 0,
            woBf, DIM_D, 0, 0, 0,
            xout, DIM_D, 0, 0, 0,
            nullptr, resid, DIM_D,
            DIM_D, 1, 0);
    };

    // ---------------- self attention ----------------
    ln_kernel<<<NROWS, blk, 0, stream>>>(x, ln1g, ln1b, hBf);
    gemm_kernel<128, 0, 1><<<gProj, blk, 0, stream>>>(hBf, DIM_D, 0,0,0, wBf[0], DIM_D, 0,0,0,
        Qbf, DIM_D, 0,0,0, nullptr, nullptr, 0, DIM_D, 1, 0);
    gemm_kernel<128, 0, 1><<<gProj, blk, 0, stream>>>(hBf, DIM_D, 0,0,0, wBf[1], DIM_D, 0,0,0,
        Kbf, DIM_D, 0,0,0, nullptr, nullptr, 0, DIM_D, 1, 0);
    gemm_kernel<128, 0, 1><<<gProj, blk, 0, stream>>>(hBf, DIM_D, 0,0,0, wBf[2], DIM_D, 0,0,0,
        Vbf, DIM_D, 0,0,0, nullptr, nullptr, 0, DIM_D, 1, 0);
    run_attention(wBf[3], x, x1);

    // ---------------- cross attention ----------------
    ln_kernel<<<NROWS, blk, 0, stream>>>(x1, ln2g, ln2b, hBf);
    gemm_kernel<128, 0, 1><<<gProj, blk, 0, stream>>>(hBf, DIM_D, 0,0,0, wBf[4], DIM_D, 0,0,0,
        Qbf, DIM_D, 0,0,0, nullptr, nullptr, 0, DIM_D, 1, 0);
    gemm_kernel<128, 0, 1><<<gProj, blk, 0, stream>>>(memBf, DIM_D, 0,0,0, wBf[5], DIM_D, 0,0,0,
        Kbf, DIM_D, 0,0,0, nullptr, nullptr, 0, DIM_D, 1, 0);
    gemm_kernel<128, 0, 1><<<gProj, blk, 0, stream>>>(memBf, DIM_D, 0,0,0, wBf[6], DIM_D, 0,0,0,
        Vbf, DIM_D, 0,0,0, nullptr, nullptr, 0, DIM_D, 1, 0);
    run_attention(wBf[7], x1, x2);

    // ---------------- MoE ----------------
    ln_kernel<<<NROWS, blk, 0, stream>>>(x2, ln3g, ln3b, hBf);
    gate_kernel<<<NROWS, blk, 0, stream>>>(hBf, Wg, prb, tw, tidx);
    aux_kernel<<<1, blk, 0, stream>>>(prb, tidx, out + TOK);      // tuple output #2
    copy_kernel<<<(unsigned)(TOK / 256), blk, 0, stream>>>(x2, out);

    for (int e = 0; e < DIM_E; ++e) {
        // hid = gelu(h @ W1[e]^T + b1[e]) -> bf16   [4096 x 4096]
        gemm_kernel<128, 1, 1><<<dim3(DIM_F / 128, NROWS / 128, 1), blk, 0, stream>>>(
            hBf, DIM_D, 0,0,0,
            w1Bf + (size_t)e * DIM_F * DIM_D, DIM_D, 0,0,0,
            hidBf, DIM_F, 0,0,0,
            b1 + (size_t)e * DIM_F, nullptr, 0,
            DIM_D, 1, 0);
        // eo = hid @ W2[e]^T + b2[e] -> fp32        [4096 x 1024]
        gemm_kernel<128, 2, 0><<<gProj, blk, 0, stream>>>(
            hidBf, DIM_F, 0,0,0,
            w2Bf + (size_t)e * DIM_D * DIM_F, DIM_F, 0,0,0,
            eo, DIM_D, 0,0,0,
            b2 + (size_t)e * DIM_D, nullptr, 0,
            DIM_F, 1, 0);
        moe_combine_kernel<<<(unsigned)(TOK / 256), blk, 0, stream>>>(eo, tw, tidx, out, e);
    }
}
